// Decoder_89859305767772
// MI455X (gfx1250) — matrix operand order repaired and checked by reading, not compiled, not run on hardware
//
#include <hip/hip_runtime.h>
#include <stdint.h>

typedef __attribute__((ext_vector_type(16))) __bf16 v16bf;
typedef __attribute__((ext_vector_type(8)))  float  v8f;

typedef unsigned short u16;

// ---- constants ---------------------------------------------------------
#define BATCH   1024
#define NIN     256
#define NHID    512
#define NOUT    256
#define BT      16          // batch rows per workgroup
#define NTHREAD 512         // 16 waves
#define ZSTR    264         // 256 + 8 halves pad (16B-aligned rows)
#define HSTR    520         // 512 + 8 halves pad
#define CSTR    520         // f32 c-state row stride

// ws layout (in halves) -- all matrices stored in swizzled fragment order
#define O_WIH0  0u                              // 2048 x 256
#define O_WHH0  (O_WIH0 + 2048u*256u)           // 2048 x 512
#define O_WIH1  (O_WHH0 + 2048u*512u)           // 2048 x 512
#define O_WHH1  (O_WIH1 + 2048u*512u)           // 2048 x 512
#define O_FCW   (O_WHH1 + 2048u*512u)           // 256 x 512
#define O_LINW  (O_FCW  + 256u*512u)            // 256 x 256
#define O_END   (O_LINW + 256u*256u)            // total halves

union FragBF { v16bf v; uint4 q[2]; };

__device__ __forceinline__ u16 f2bf(float x) {
    union { float f; uint32_t u; } c; c.f = x;
    uint32_t u = c.u;
    u += 0x7FFFu + ((u >> 16) & 1u);           // round-to-nearest-even
    return (u16)(u >> 16);
}

__device__ __forceinline__ float sigf(float x) {
    return 1.0f / (1.0f + __expf(-x));
}
__device__ __forceinline__ float tanhf_(float x) {
    float e = __expf(-2.0f * fabsf(x));        // e in (0,1], no inf
    float t = (1.0f - e) / (1.0f + e);
    return copysignf(t, x);
}

// A fragment (ISA 7.12.2, 16-bit 16x32): lane m=l%16, kh=l/16
// elems 0-7 = X[m][kb+8*kh .. +7], elems 8-15 = X[m][kb+16+8*kh .. +7]
__device__ __forceinline__ v16bf load_a(const u16* row, int kb, int kh) {
    FragBF f;
    f.q[0] = *reinterpret_cast<const uint4*>(row + kb + 8 * kh);
    f.q[1] = *reinterpret_cast<const uint4*>(row + kb + 16 + 8 * kh);
    return f.v;
}

// B fragment from swizzled weights: p already includes lane*16 offset;
// each (tile,kstep) block is 1KB contiguous, lane l reads 32B at l*32.
__device__ __forceinline__ v16bf load_bs(const u16* p) {
    FragBF f;
    f.q[0] = *reinterpret_cast<const uint4*>(p);
    f.q[1] = *reinterpret_cast<const uint4*>(p + 8);
    return f.v;
}

#define WMMA_BF16(A, B, C) \
    __builtin_amdgcn_wmma_f32_16x16x32_bf16(false, (A), false, (B), (short)0, (C), false, false)

// one 16-col hidden tile: 4 gate GEMMs + LSTM cell; c-state f32 in LDS
__device__ __forceinline__ void lstm_gates_tile(
    const u16* __restrict__ x_lds, int xstr, int kx,   // layer input activations
    const u16* __restrict__ h_lds,                     // prev hidden (K=512, HSTR)
    const u16* __restrict__ wih,                       // swizzled [2048, kx]
    const u16* __restrict__ whh,                       // swizzled [2048, 512]
    const float* __restrict__ bsum,                    // b_ih + b_hh  [2048]
    int ht, int lane, int lane16, int kh,
    float* __restrict__ cst,                           // LDS f32 c-state (CSTR)
    u16* __restrict__ hout)
{
    const int colb = ht * 16 + lane16;
    v8f ai, af, ag, ao;
    {
        float bi = bsum[colb], bf = bsum[512 + colb];
        float bg = bsum[1024 + colb], bo = bsum[1536 + colb];
#pragma unroll
        for (int r = 0; r < 8; ++r) { ai[r] = bi; af[r] = bf; ag[r] = bg; ao[r] = bo; }
    }

    // x @ W_ih^T
    {
        const int ks = kx >> 5;
        const u16* ar = x_lds + lane16 * xstr;
        const u16* w0 = wih + ((size_t)(0 * 32 + ht) * ks) * 512 + lane * 16;
        const u16* w1 = wih + ((size_t)(1 * 32 + ht) * ks) * 512 + lane * 16;
        const u16* w2 = wih + ((size_t)(2 * 32 + ht) * ks) * 512 + lane * 16;
        const u16* w3 = wih + ((size_t)(3 * 32 + ht) * ks) * 512 + lane * 16;
#pragma unroll 1
        for (int k = 0; k < ks; ++k) {
            v16bf a = load_a(ar, k * 32, kh);
            ai = WMMA_BF16(a, load_bs(w0), ai); w0 += 512;
            af = WMMA_BF16(a, load_bs(w1), af); w1 += 512;
            ag = WMMA_BF16(a, load_bs(w2), ag); w2 += 512;
            ao = WMMA_BF16(a, load_bs(w3), ao); w3 += 512;
        }
    }
    // h_prev @ W_hh^T  (K = 512 -> 16 k-steps)
    {
        const u16* ar = h_lds + lane16 * HSTR;
        const u16* w0 = whh + ((size_t)(0 * 32 + ht) * 16) * 512 + lane * 16;
        const u16* w1 = whh + ((size_t)(1 * 32 + ht) * 16) * 512 + lane * 16;
        const u16* w2 = whh + ((size_t)(2 * 32 + ht) * 16) * 512 + lane * 16;
        const u16* w3 = whh + ((size_t)(3 * 32 + ht) * 16) * 512 + lane * 16;
#pragma unroll 1
        for (int k = 0; k < 16; ++k) {
            v16bf a = load_a(ar, k * 32, kh);
            ai = WMMA_BF16(a, load_bs(w0), ai); w0 += 512;
            af = WMMA_BF16(a, load_bs(w1), af); w1 += 512;
            ag = WMMA_BF16(a, load_bs(w2), ag); w2 += 512;
            ao = WMMA_BF16(a, load_bs(w3), ao); w3 += 512;
        }
    }
    // elementwise cell; C layout: vgpr r, lane -> (m = r+8*kh, n = colb)
#pragma unroll
    for (int r = 0; r < 8; ++r) {
        const int m = r + 8 * kh;
        float iv = sigf(ai[r]), fv = sigf(af[r]);
        float gv = tanhf_(ag[r]), ov = sigf(ao[r]);
        float cn = fv * cst[m * CSTR + colb] + iv * gv;
        cst[m * CSTR + colb] = cn;
        hout[m * HSTR + colb] = f2bf(ov * tanhf_(cn));
    }
}

__device__ __forceinline__ v8f dense_tile(
    const u16* __restrict__ x_lds, int xstr, int K,
    const u16* __restrict__ w, int nt,                 // swizzled weights, n-tile
    float bias, int lane, int lane16, int kh)
{
    v8f acc;
#pragma unroll
    for (int r = 0; r < 8; ++r) acc[r] = bias;
    const int ks = K >> 5;
    const u16* ar = x_lds + lane16 * xstr;
    const u16* wp = w + ((size_t)nt * ks) * 512 + lane * 16;
#pragma unroll 1
    for (int k = 0; k < ks; ++k) {
        v16bf a = load_a(ar, k * 32, kh);
        acc = WMMA_BF16(a, load_bs(wp), acc); wp += 512;
    }
    return acc;
}

// ---- prep kernels ------------------------------------------------------
// fp32 row-major [nrows, K]  ->  bf16 swizzled fragment order:
// dst[(((nt*(K/32) + kb)*32 + lane)*16 + e)] = W[nt*16 + lane%16][kb*32 + (e/8)*16 + (lane/16)*8 + e%8]
__global__ void k_swz_bf16(const float* __restrict__ s, u16* __restrict__ d,
                           int nrows, int K) {
    int i = blockIdx.x * blockDim.x + threadIdx.x;
    if (i >= nrows * K) return;
    int e   = i & 15;
    int l   = (i >> 4) & 31;
    int blk = i >> 9;
    int ks  = K >> 5;
    int kb  = blk % ks;
    int nt  = blk / ks;
    int row = nt * 16 + (l & 15);
    int col = kb * 32 + (e & 8) * 2 + (l >> 4) * 8 + (e & 7);
    d[i] = f2bf(s[(size_t)row * K + col]);
}

__global__ void k_bias_sum(const float* a0, const float* b0, float* o0,
                           const float* a1, const float* b1, float* o1, int n) {
    int i = blockIdx.x * blockDim.x + threadIdx.x;
    if (i < n) { o0[i] = a0[i] + b0[i]; o1[i] = a1[i] + b1[i]; }
}

// ---- persistent decoder kernel ----------------------------------------
__global__ __launch_bounds__(NTHREAD) void k_lstm_decoder(
    const float* __restrict__ z0, const float* __restrict__ h0in, const float* __restrict__ c0in,
    const u16* __restrict__ wih0, const u16* __restrict__ whh0,
    const u16* __restrict__ wih1, const u16* __restrict__ whh1,
    const u16* __restrict__ fcw,  const u16* __restrict__ linw,
    const float* __restrict__ bsum0, const float* __restrict__ bsum1,
    const float* __restrict__ fcb,   const float* __restrict__ linb,
    float* __restrict__ out, int T2)
{
    __shared__ __align__(16) u16   sZ[BT * ZSTR];
    __shared__ __align__(16) u16   sH0[2][BT * HSTR];
    __shared__ __align__(16) u16   sH1[2][BT * HSTR];
    __shared__ __align__(16) float sC0[BT * CSTR];
    __shared__ __align__(16) float sC1[BT * CSTR];

    const int tid    = threadIdx.x;
    const int wave   = tid >> 5;          // 0..15
    const int lane   = tid & 31;
    const int lane16 = lane & 15;
    const int kh     = lane >> 4;
    const int bbase  = blockIdx.x * BT;

    // --- init LDS activations (fp32 -> bf16) and f32 c-state ---
    for (int idx = tid; idx < BT * NIN; idx += NTHREAD) {
        int r = idx >> 8, c = idx & 255;
        sZ[r * ZSTR + c] = f2bf(z0[(size_t)(bbase + r) * NIN + c]);
    }
    for (int idx = tid; idx < BT * NHID; idx += NTHREAD) {
        int r = idx >> 9, c = idx & 511;
        sH0[0][r * HSTR + c] = f2bf(h0in[(size_t)(bbase + r) * NHID + c]);
        sH1[0][r * HSTR + c] = f2bf(h0in[(size_t)(BATCH + bbase + r) * NHID + c]);
        sC0[r * CSTR + c] = c0in[(size_t)(bbase + r) * NHID + c];
        sC1[r * CSTR + c] = c0in[(size_t)(BATCH + bbase + r) * NHID + c];
    }
    __syncthreads();

    int cur = 0;
    for (int t = 0; t < T2; ++t) {
        int nxt = cur ^ 1;
        u16* h0c = sH0[cur]; u16* h0n = sH0[nxt];
        u16* h1c = sH1[cur]; u16* h1n = sH1[nxt];

        // ---- layer 0: gates = z @ Wih0^T + h0 @ Whh0^T + bsum0 ----
#pragma unroll 1
        for (int j = 0; j < 2; ++j) {
            lstm_gates_tile(sZ, ZSTR, NIN, h0c, wih0, whh0, bsum0,
                            wave + 16 * j, lane, lane16, kh, sC0, h0n);
        }
        __syncthreads();

        // ---- layer 1: gates = h0_new @ Wih1^T + h1 @ Whh1^T + bsum1 ----
#pragma unroll 1
        for (int j = 0; j < 2; ++j) {
            lstm_gates_tile(h0n, HSTR, NHID, h1c, wih1, whh1, bsum1,
                            wave + 16 * j, lane, lane16, kh, sC1, h1n);
        }
        __syncthreads();

        // ---- fc feedback: z_new = h1_new @ fcW^T + fcb -> sZ ----
        {
            int nt = wave;                 // 16 tiles, one per wave
            int colb = nt * 16 + lane16;
            v8f zt = dense_tile(h1n, HSTR, NHID, fcw, nt, fcb[colb],
                                lane, lane16, kh);
#pragma unroll
            for (int r = 0; r < 8; ++r)
                sZ[(r + 8 * kh) * ZSTR + colb] = f2bf(zt[r]);
        }
        __syncthreads();

        // ---- output head: out[t] = z_new @ linW^T + linb ----
        {
            int nt = wave;                 // 16 tiles, one per wave
            int colb = nt * 16 + lane16;
            v8f ot = dense_tile(sZ, ZSTR, NIN, linw, nt, linb[colb],
                                lane, lane16, kh);
#pragma unroll
            for (int r = 0; r < 8; ++r)
                __builtin_nontemporal_store(
                    ot[r],
                    &out[((size_t)t * BATCH + bbase + r + 8 * kh) * NOUT + colb]);
        }
        __syncthreads();

        cur = nxt;
    }
}

// ---- host launcher -----------------------------------------------------
extern "C" void kernel_launch(void* const* d_in, const int* in_sizes, int n_in,
                              void* d_out, int out_size, void* d_ws, size_t ws_size,
                              hipStream_t stream) {
    const float* z0   = (const float*)d_in[0];
    const float* h0   = (const float*)d_in[1];
    const float* c0   = (const float*)d_in[2];
    const float* Wih0 = (const float*)d_in[3];
    const float* Whh0 = (const float*)d_in[4];
    const float* bih0 = (const float*)d_in[5];
    const float* bhh0 = (const float*)d_in[6];
    const float* Wih1 = (const float*)d_in[7];
    const float* Whh1 = (const float*)d_in[8];
    const float* bih1 = (const float*)d_in[9];
    const float* bhh1 = (const float*)d_in[10];
    const float* fcW  = (const float*)d_in[11];
    const float* fcb  = (const float*)d_in[12];
    const float* linW = (const float*)d_in[13];
    const float* linb = (const float*)d_in[14];
    float* out = (float*)d_out;

    const int T2 = out_size / (BATCH * NOUT);

    u16*   ws  = (u16*)d_ws;
    float* bs0 = (float*)(ws + O_END);
    float* bs1 = bs0 + 2048;

    struct { const float* s; u16* d; int nrows; int K; } cv[6] = {
        { Wih0, ws + O_WIH0, 2048, 256 },
        { Whh0, ws + O_WHH0, 2048, 512 },
        { Wih1, ws + O_WIH1, 2048, 512 },
        { Whh1, ws + O_WHH1, 2048, 512 },
        { fcW,  ws + O_FCW,  256,  512 },
        { linW, ws + O_LINW, 256,  256 },
    };
    for (int i = 0; i < 6; ++i) {
        int n = cv[i].nrows * cv[i].K;
        k_swz_bf16<<<(n + 255) / 256, 256, 0, stream>>>(cv[i].s, cv[i].d,
                                                        cv[i].nrows, cv[i].K);
    }
    k_bias_sum<<<(2048 + 255) / 256, 256, 0, stream>>>(bih0, bhh0, bs0, bih1, bhh1, bs1, 2048);

    k_lstm_decoder<<<BATCH / BT, NTHREAD, 0, stream>>>(
        z0, h0, c0,
        ws + O_WIH0, ws + O_WHH0, ws + O_WIH1, ws + O_WHH1,
        ws + O_FCW, ws + O_LINW,
        bs0, bs1, fcb, linb, out, T2);
}
